// TransformerDecoder_76330158784753
// MI455X (gfx1250) — matrix operand order repaired and checked by reading, not compiled, not run on hardware
//
#include <hip/hip_runtime.h>

typedef __attribute__((ext_vector_type(16))) __bf16 v16bf;
typedef __attribute__((ext_vector_type(8)))  float  v8f;

#define D_MODEL 512
#define NHEADS  8
#define DHEAD   64
#define DFF_    2048
#define NL_     6
#define SEQ     512
#define BATCH   8
#define MROWS   (SEQ*BATCH)   /* 4096 */

#define WMMA_BF16(a, b, c) \
  __builtin_amdgcn_wmma_f32_16x16x32_bf16(false, (a), false, (b), (short)0, (c), false, false)

// GEMM output modes
#define OUT_F32        0
#define OUT_F32_RELU   1
#define OUT_BF16       2
#define OUT_BF16_VT    3   // bf16, stored transposed per head: Vt[b][h][dim][key]

// ---------------------------------------------------------------------------
// Kernel 1: embedding gather * sqrt(D) + sinusoidal positional encoding
// ---------------------------------------------------------------------------
__global__ __launch_bounds__(256) void embed_pe_kernel(
    const int* __restrict__ targets, const float* __restrict__ emb,
    float* __restrict__ X)
{
  int idx = blockIdx.x * 256 + threadIdx.x;        // 0 .. MROWS*D-1
  int d   = idx & (D_MODEL - 1);
  int sb  = idx >> 9;                              // row = b*SEQ + s
  int b   = sb >> 9;
  int s   = sb & (SEQ - 1);
  int t   = targets[s * BATCH + b];                // targets is (S,B)
  int i2  = d & ~1;
  float freq = __expf(-(float)i2 * (9.210340371976184f / 512.0f)); // ln(1e4)/D
  float ang  = (float)s * freq;
  float pe   = (d & 1) ? __cosf(ang) : __sinf(ang);
  X[idx] = emb[(size_t)t * D_MODEL + d] * 22.627416997969522f + pe;
}

// ---------------------------------------------------------------------------
// Kernel 2: C = act(A[M,K] @ W[K,N] + bias[N]) via bf16 WMMA, f32 accumulate.
// Block tile 128x128, K-step 64, 256 threads = 8 waves (2M x 4N),
// wave tile 64x32 -> 16 WMMAs per staging round.
// Software-pipelined: tiles for step k+1 are loaded into registers while the
// WMMAs of step k run. bf16-A staging uses GLOBAL_LOAD_ASYNC_TO_LDS_B128.
// ---------------------------------------------------------------------------
#define BM 128
#define BN 128
#define BK 64
#define LDA_S 72   // 64 + 8 pad (bf16 elems); row = 144 B (16B multiple)
#define LDB_S 72

template <int MODE, typename TA>
__global__ __launch_bounds__(256) void gemm_kernel(
    const TA* __restrict__ A, const float* __restrict__ W,
    const float* __restrict__ bias, void* __restrict__ Cv,
    int M, int N, int K)
{
  __shared__ __align__(16) __bf16 As[BM * LDA_S];
  __shared__ __align__(16) __bf16 Bs[BN * LDB_S];   // transposed: [n][k]

  const int tid  = threadIdx.x;
  const int lane = tid & 31;
  const int wave = tid >> 5;
  const int wm   = wave >> 2;   // 0..1
  const int wn   = wave & 3;    // 0..3
  const int bm0  = blockIdx.x * BM;
  const int bn0  = blockIdx.y * BN;
  const int nn   = lane & 15;
  const int g    = lane >> 4;

  v8f acc[4][2];
#pragma unroll
  for (int i = 0; i < 4; ++i)
#pragma unroll
    for (int j = 0; j < 2; ++j)
#pragma unroll
      for (int e = 0; e < 8; ++e) acc[i][j][e] = 0.0f;

  float4 wreg[8];
  float4 areg[8];

  // --- register-prefetch helpers ---
  auto loadW = [&](int k0) {
#pragma unroll
    for (int j = 0; j < 8; ++j) {
      int f4   = tid + j * 256;          // 2048 float4 (64 x 128)
      int krow = f4 >> 5;                // 32 float4 per k-row
      int n4   = (f4 & 31) * 4;
      wreg[j]  = *(const float4*)(W + (size_t)(k0 + krow) * N + bn0 + n4);
    }
  };
  auto loadA32 = [&](int k0) {
#pragma unroll
    for (int j = 0; j < 8; ++j) {
      int f4   = tid + j * 256;          // 2048 float4 (128 x 64)
      int row  = f4 >> 4;                // 16 float4 per row
      int col4 = (f4 & 15) * 4;
      areg[j]  = *(const float4*)((const float*)A + (size_t)(bm0 + row) * K + k0 + col4);
    }
  };
  // bf16 A: pure byte copy -> async global->LDS (CDNA5), 4x16B per thread
  auto asyncA16 = [&](int k0) {
#pragma unroll
    for (int j = 0; j < 4; ++j) {
      int c8   = tid + j * 256;          // 1024 16B chunks
      int row  = c8 >> 3;                // 8 chunks per row
      int col8 = (c8 & 7) * 8;
      unsigned ldsAddr = (unsigned)(uintptr_t)&As[row * LDA_S + col8];
      const __bf16* gsrc = (const __bf16*)A + (size_t)(bm0 + row) * K + k0 + col8;
      asm volatile("global_load_async_to_lds_b128 %0, %1, off"
                   :: "v"(ldsAddr), "v"(gsrc) : "memory");
    }
  };

  // prologue
  loadW(0);
  if constexpr (sizeof(TA) == 4) loadA32(0);

  for (int k0 = 0; k0 < K; k0 += BK) {
    if constexpr (sizeof(TA) == 2) asyncA16(k0);

    // commit W regs -> Bs transposed [n][k], f32 -> bf16
#pragma unroll
    for (int j = 0; j < 8; ++j) {
      int f4   = tid + j * 256;
      int krow = f4 >> 5;
      int n4   = (f4 & 31) * 4;
      Bs[(n4 + 0) * LDB_S + krow] = (__bf16)wreg[j].x;
      Bs[(n4 + 1) * LDB_S + krow] = (__bf16)wreg[j].y;
      Bs[(n4 + 2) * LDB_S + krow] = (__bf16)wreg[j].z;
      Bs[(n4 + 3) * LDB_S + krow] = (__bf16)wreg[j].w;
    }
    if constexpr (sizeof(TA) == 4) {
      // commit A regs -> As row-major, f32 -> bf16
#pragma unroll
      for (int j = 0; j < 8; ++j) {
        int f4   = tid + j * 256;
        int row  = f4 >> 4;
        int col4 = (f4 & 15) * 4;
        __bf16* dst = &As[row * LDA_S + col4];
        dst[0] = (__bf16)areg[j].x; dst[1] = (__bf16)areg[j].y;
        dst[2] = (__bf16)areg[j].z; dst[3] = (__bf16)areg[j].w;
      }
    } else {
      asm volatile("s_wait_asynccnt 0x0" ::: "memory");
    }
    __syncthreads();

    // prefetch next tiles into registers (overlaps with WMMAs below)
    const bool more = (k0 + BK) < K;
    if (more) {
      loadW(k0 + BK);
      if constexpr (sizeof(TA) == 4) loadA32(k0 + BK);
    }

#pragma unroll
    for (int kc = 0; kc < 2; ++kc) {     // two 32-wide K chunks
      v16bf bfrag[2];
#pragma unroll
      for (int nf = 0; nf < 2; ++nf) {
        const __bf16* src = &Bs[(wn * 32 + nf * 16 + nn) * LDB_S + kc * 32 + g * 16];
#pragma unroll
        for (int e = 0; e < 16; ++e) bfrag[nf][e] = src[e];
      }
#pragma unroll
      for (int ms = 0; ms < 4; ++ms) {
        const __bf16* arow = &As[(wm * 64 + ms * 16 + nn) * LDA_S + kc * 32];
        v16bf afrag;
#pragma unroll
        for (int v = 0; v < 8; ++v) {
          int kk = g * 16 + v * 2;
          afrag[2 * v]     = arow[kk];
          afrag[2 * v + 1] = arow[kk + 1];
        }
        acc[ms][0] = WMMA_BF16(afrag, bfrag[0], acc[ms][0]);
        acc[ms][1] = WMMA_BF16(afrag, bfrag[1], acc[ms][1]);
      }
    }
    __syncthreads();
  }

  // ---- epilogue ----
#pragma unroll
  for (int nf = 0; nf < 2; ++nf) {
    const int gcol = bn0 + wn * 32 + nf * 16 + nn;
    const float bv = bias ? bias[gcol] : 0.0f;
#pragma unroll
    for (int ms = 0; ms < 4; ++ms)
#pragma unroll
      for (int v = 0; v < 8; ++v) {
        const int row = bm0 + wm * 64 + ms * 16 + v + 8 * g;
        float val = acc[ms][nf][v] + bv;
        if constexpr (MODE == OUT_F32_RELU) val = fmaxf(val, 0.0f);
        if constexpr (MODE == OUT_F32 || MODE == OUT_F32_RELU) {
          ((float*)Cv)[(size_t)row * N + gcol] = val;
        } else if constexpr (MODE == OUT_BF16) {
          ((__bf16*)Cv)[(size_t)row * N + gcol] = (__bf16)val;
        } else {  // OUT_BF16_VT : Vt[b][h][dim][key]
          const int b   = row >> 9;            // /SEQ
          const int key = row & (SEQ - 1);
          const int h   = gcol >> 6;           // /DHEAD
          const int dim = gcol & (DHEAD - 1);
          ((__bf16*)Cv)[(((size_t)b * NHEADS + h) * DHEAD + dim) * SEQ + key] =
              (__bf16)val;
        }
      }
  }
}

// ---------------------------------------------------------------------------
// Kernel 3: flash attention, bf16 in / bf16 out. 1 wave per 16 q-rows.
// Q,K: (B*S, D_MODEL) bf16, head h at cols [h*64, h*64+64).
// V:   transposed bf16 Vt[b][h][dim][key] -> contiguous fragment reads.
// ---------------------------------------------------------------------------
__device__ inline float red_max16(float x) {
#pragma unroll
  for (int off = 8; off > 0; off >>= 1) x = fmaxf(x, __shfl_xor(x, off, 16));
  return x;
}
__device__ inline float red_sum16(float x) {
#pragma unroll
  for (int off = 8; off > 0; off >>= 1) x += __shfl_xor(x, off, 16);
  return x;
}

__global__ __launch_bounds__(32) void attn_kernel(
    const __bf16* __restrict__ Qm, const __bf16* __restrict__ Km,
    const __bf16* __restrict__ Vt, __bf16* __restrict__ Om,
    int causal, int Skv)
{
  __shared__ __align__(16) __bf16 Pst[16 * 36];   // P tile 16x32 (+pad)

  const int lane = threadIdx.x;
  const int nn   = lane & 15;
  const int g    = lane >> 4;
  const int qt   = blockIdx.x & (SEQ / 16 - 1);
  const int bh   = blockIdx.x >> 5;               // /(SEQ/16)
  const int h    = bh & (NHEADS - 1);
  const int b    = bh >> 3;
  const int q0   = qt * 16;

  const __bf16* qbase = Qm + (size_t)b * SEQ * D_MODEL + h * DHEAD;
  const __bf16* kbase = Km + (size_t)b * SEQ * D_MODEL + h * DHEAD;
  const __bf16* vbase = Vt + (size_t)(b * NHEADS + h) * DHEAD * SEQ;

  // Q A-fragments (16x32 each, dh 0..31 / 32..63); row = nn, K-half = g
  v16bf qa0, qa1;
  {
    const __bf16* qrow = qbase + (size_t)(q0 + nn) * D_MODEL;
#pragma unroll
    for (int v = 0; v < 8; ++v) {
      int kk = g * 16 + v * 2;
      qa0[2 * v]     = qrow[kk];
      qa0[2 * v + 1] = qrow[kk + 1];
      qa1[2 * v]     = qrow[32 + kk];
      qa1[2 * v + 1] = qrow[32 + kk + 1];
    }
  }

  float mrow[8], lrow[8];
  v8f acc[4];
#pragma unroll
  for (int v = 0; v < 8; ++v) { mrow[v] = -3.0e38f; lrow[v] = 0.0f; }
#pragma unroll
  for (int t = 0; t < 4; ++t)
#pragma unroll
    for (int v = 0; v < 8; ++v) acc[t][v] = 0.0f;

  const int nch = causal ? (qt / 2 + 1) : (Skv / 32);

  for (int c = 0; c < nch; ++c) {
    const int kb = c * 32;

    // scores: 16q x 32key as two 16x16 C tiles
    v8f s[2];
#pragma unroll
    for (int t = 0; t < 2; ++t) {
#pragma unroll
      for (int v = 0; v < 8; ++v) s[t][v] = 0.0f;
      const __bf16* krow = kbase + (size_t)(kb + t * 16 + nn) * D_MODEL;
      v16bf kf0, kf1;
#pragma unroll
      for (int e = 0; e < 16; ++e) {
        kf0[e] = krow[g * 16 + e];        // dh 0..31 block
        kf1[e] = krow[32 + g * 16 + e];   // dh 32..63 block
      }
      s[t] = WMMA_BF16(qa0, kf0, s[t]);
      s[t] = WMMA_BF16(qa1, kf1, s[t]);
    }

    // scale + causal mask; online softmax
#pragma unroll
    for (int v = 0; v < 8; ++v) {
      const int row = v + 8 * g;
#pragma unroll
      for (int t = 0; t < 2; ++t) {
        float sv = s[t][v] * 0.125f;      // 1/sqrt(64)
        if (causal && (kb + t * 16 + nn) > (q0 + row)) sv = -1.0e30f;
        s[t][v] = sv;
      }
      float rm = red_max16(fmaxf(s[0][v], s[1][v]));
      float mn = fmaxf(mrow[v], rm);
      float al = __expf(mrow[v] - mn);
      mrow[v]  = mn;
      s[0][v]  = __expf(s[0][v] - mn);
      s[1][v]  = __expf(s[1][v] - mn);
      lrow[v]  = lrow[v] * al + red_sum16(s[0][v] + s[1][v]);
#pragma unroll
      for (int t = 0; t < 4; ++t) acc[t][v] *= al;
    }

    // stage P (16x32) to LDS, reload as A-fragment
#pragma unroll
    for (int v = 0; v < 8; ++v) {
      Pst[(v + 8 * g) * 36 + nn]      = (__bf16)s[0][v];
      Pst[(v + 8 * g) * 36 + 16 + nn] = (__bf16)s[1][v];
    }
    asm volatile("s_wait_dscnt 0" ::: "memory");
    v16bf pf;
#pragma unroll
    for (int v = 0; v < 8; ++v) {
      int kk = g * 16 + v * 2;
      pf[2 * v]     = Pst[nn * 36 + kk];
      pf[2 * v + 1] = Pst[nn * 36 + kk + 1];
    }

    // acc += P @ V ; Vt rows are contiguous in key -> vector loads
#pragma unroll
    for (int t = 0; t < 4; ++t) {
      const __bf16* vrow = vbase + (size_t)(t * 16 + nn) * SEQ + kb + g * 16;
      v16bf vf;
#pragma unroll
      for (int e = 0; e < 16; ++e) vf[e] = vrow[e];
      acc[t] = WMMA_BF16(pf, vf, acc[t]);
    }
  }

  // normalize and store ctx (bf16)
#pragma unroll
  for (int t = 0; t < 4; ++t)
#pragma unroll
    for (int v = 0; v < 8; ++v) {
      int row = v + 8 * g;
      Om[(size_t)(b * SEQ + q0 + row) * D_MODEL + h * DHEAD + t * 16 + nn] =
          (__bf16)(acc[t][v] / lrow[v]);
    }
}

// ---------------------------------------------------------------------------
// Kernel 4: Out[row] = LayerNorm(X[row] + Y[row]) * g + b   (D = 512)
// ---------------------------------------------------------------------------
__global__ __launch_bounds__(256) void add_ln_kernel(
    const float* __restrict__ X, const float* __restrict__ Y,
    const float* __restrict__ gw, const float* __restrict__ bw,
    float* __restrict__ Out)
{
  const int row  = blockIdx.x;
  const int tid  = threadIdx.x;
  const int lane = tid & 31;
  const int wave = tid >> 5;
  __shared__ float ssum[8], ssq[8], mv[2];

  const size_t base = (size_t)row * D_MODEL;
  float v0 = X[base + tid]       + Y[base + tid];
  float v1 = X[base + 256 + tid] + Y[base + 256 + tid];
  float s  = v0 + v1;
  float sq = v0 * v0 + v1 * v1;
#pragma unroll
  for (int off = 16; off > 0; off >>= 1) {
    s  += __shfl_xor(s,  off, 32);
    sq += __shfl_xor(sq, off, 32);
  }
  if (lane == 0) { ssum[wave] = s; ssq[wave] = sq; }
  __syncthreads();
  if (tid == 0) {
    float ts = 0.f, tq = 0.f;
#pragma unroll
    for (int i = 0; i < 8; ++i) { ts += ssum[i]; tq += ssq[i]; }
    float mean = ts / 512.0f;
    mv[0] = mean;
    mv[1] = rsqrtf(tq / 512.0f - mean * mean + 1e-5f);
  }
  __syncthreads();
  const float mean = mv[0], inv = mv[1];
  Out[base + tid]       = (v0 - mean) * inv * gw[tid]       + bw[tid];
  Out[base + 256 + tid] = (v1 - mean) * inv * gw[256 + tid] + bw[256 + tid];
}

// ---------------------------------------------------------------------------
// Host launcher
// ---------------------------------------------------------------------------
extern "C" void kernel_launch(void* const* d_in, const int* in_sizes, int n_in,
                              void* d_out, int out_size, void* d_ws, size_t ws_size,
                              hipStream_t stream)
{
  const int*   targets = (const int*)  d_in[0];
  const float* memory  = (const float*)d_in[1];
  // d_in[2], d_in[3]: trg_mask (triu) / memory_mask (all false) — done analytically
  const float* emb     = (const float*)d_in[4];
  const float* Wq_s = (const float*)d_in[5],  *Wk_s = (const float*)d_in[6];
  const float* Wv_s = (const float*)d_in[7],  *Wo_s = (const float*)d_in[8];
  const float* Wq_c = (const float*)d_in[9],  *Wk_c = (const float*)d_in[10];
  const float* Wv_c = (const float*)d_in[11], *Wo_c = (const float*)d_in[12];
  const float* bq_s = (const float*)d_in[13], *bk_s = (const float*)d_in[14];
  const float* bv_s = (const float*)d_in[15], *bo_s = (const float*)d_in[16];
  const float* bq_c = (const float*)d_in[17], *bk_c = (const float*)d_in[18];
  const float* bv_c = (const float*)d_in[19], *bo_c = (const float*)d_in[20];
  const float* W1   = (const float*)d_in[21], *b1   = (const float*)d_in[22];
  const float* W2   = (const float*)d_in[23], *b2   = (const float*)d_in[24];
  const float* g1   = (const float*)d_in[25], *g2   = (const float*)d_in[26];
  const float* g3   = (const float*)d_in[27];
  const float* be1  = (const float*)d_in[28], *be2  = (const float*)d_in[29];
  const float* be3  = (const float*)d_in[30];

  const size_t ND = (size_t)MROWS * D_MODEL;     // 2M elems
  const size_t NF = (size_t)MROWS * DFF_;        // 8M elems
  char* wsb = (char*)d_ws;
  float*  X    = (float*) (wsb);                          // 8 MB
  float*  T1   = (float*) (wsb + ND * 4);                 // 8 MB
  float*  F1   = (float*) (wsb + 2 * ND * 4);             // 32 MB
  __bf16* Qb   = (__bf16*)(wsb + 2 * ND * 4 + NF * 4);    // 4 MB
  __bf16* Kbf  = (__bf16*)(wsb + 2 * ND * 4 + NF * 4 + 1 * ND * 2);
  __bf16* Vtb  = (__bf16*)(wsb + 2 * ND * 4 + NF * 4 + 2 * ND * 2);
  __bf16* CTXb = (__bf16*)(wsb + 2 * ND * 4 + NF * 4 + 3 * ND * 2);

  const dim3 blk256(256), blk32(32);
  const dim3 gridD(MROWS / BM, D_MODEL / BN);    // N = 512  -> (32, 4)
  const dim3 gridF(MROWS / BM, DFF_ / BN);       // N = 2048 -> (32, 16)
  const int  attnGrid = BATCH * NHEADS * (SEQ / 16);  // 2048

  embed_pe_kernel<<<(MROWS * D_MODEL) / 256, blk256, 0, stream>>>(targets, emb, X);

  for (int l = 0; l < NL_; ++l) {
    const size_t wDD = (size_t)l * D_MODEL * D_MODEL;
    const size_t wDF = (size_t)l * D_MODEL * DFF_;
    const size_t vD  = (size_t)l * D_MODEL;
    const size_t vF  = (size_t)l * DFF_;

    // --- self attention ---
    gemm_kernel<OUT_BF16, float><<<gridD, blk256, 0, stream>>>(X, Wq_s + wDD, bq_s + vD, Qb,  MROWS, D_MODEL, D_MODEL);
    gemm_kernel<OUT_BF16, float><<<gridD, blk256, 0, stream>>>(X, Wk_s + wDD, bk_s + vD, Kbf, MROWS, D_MODEL, D_MODEL);
    gemm_kernel<OUT_BF16_VT, float><<<gridD, blk256, 0, stream>>>(X, Wv_s + wDD, bv_s + vD, Vtb, MROWS, D_MODEL, D_MODEL);
    attn_kernel<<<attnGrid, blk32, 0, stream>>>(Qb, Kbf, Vtb, CTXb, /*causal=*/1, SEQ);
    gemm_kernel<OUT_F32, __bf16><<<gridD, blk256, 0, stream>>>(CTXb, Wo_s + wDD, bo_s + vD, T1, MROWS, D_MODEL, D_MODEL);
    add_ln_kernel<<<MROWS, blk256, 0, stream>>>(X, T1, g1 + vD, be1 + vD, X);

    // --- cross attention ---
    gemm_kernel<OUT_BF16, float><<<gridD, blk256, 0, stream>>>(X,      Wq_c + wDD, bq_c + vD, Qb,  MROWS, D_MODEL, D_MODEL);
    gemm_kernel<OUT_BF16, float><<<gridD, blk256, 0, stream>>>(memory, Wk_c + wDD, bk_c + vD, Kbf, MROWS, D_MODEL, D_MODEL);
    gemm_kernel<OUT_BF16_VT, float><<<gridD, blk256, 0, stream>>>(memory, Wv_c + wDD, bv_c + vD, Vtb, MROWS, D_MODEL, D_MODEL);
    attn_kernel<<<attnGrid, blk32, 0, stream>>>(Qb, Kbf, Vtb, CTXb, /*causal=*/0, SEQ);
    gemm_kernel<OUT_F32, __bf16><<<gridD, blk256, 0, stream>>>(CTXb, Wo_c + wDD, bo_c + vD, T1, MROWS, D_MODEL, D_MODEL);
    add_ln_kernel<<<MROWS, blk256, 0, stream>>>(X, T1, g2 + vD, be2 + vD, X);

    // --- FFN ---
    gemm_kernel<OUT_F32_RELU, float><<<gridF, blk256, 0, stream>>>(X,  W1 + wDF, b1 + vF, F1, MROWS, DFF_,    D_MODEL);
    gemm_kernel<OUT_F32, float><<<gridD, blk256, 0, stream>>>(F1, W2 + wDF, b2 + vD, T1, MROWS, D_MODEL, DFF_);

    float* lnOut = (l == NL_ - 1) ? (float*)d_out : X;
    add_ln_kernel<<<MROWS, blk256, 0, stream>>>(X, T1, g3 + vD, be3 + vD, lnOut);
  }
}
